// MultiHeadAttention_87024627351855
// MI455X (gfx1250) — compile-verified, matrix operand order checked by reading, not run on hardware
//
#include <hip/hip_runtime.h>
#include <hip/hip_bf16.h>
#include <stdint.h>

typedef __attribute__((ext_vector_type(16))) __bf16 v16bf;
typedef __attribute__((ext_vector_type(8)))  __bf16 v8bf;   // 16 bytes
typedef __attribute__((ext_vector_type(8)))  float  v8f;
typedef __attribute__((ext_vector_type(4)))  unsigned int u32x4;
typedef __attribute__((ext_vector_type(8)))  int          i32x8;
typedef __attribute__((ext_vector_type(4)))  int          i32x4;

#define TSEQ   2048
#define NUNITS 1024
#define NHEADS 16
#define DHEAD  64
#define NREL   41
#define BROWS  4096   // B*T
#define NEG_BIG (-3.0e38f)

#if defined(__gfx1250__) && __has_builtin(__builtin_amdgcn_tensor_load_to_lds) && \
    __has_builtin(__builtin_amdgcn_s_wait_tensorcnt)
#define CDNA5_TDM 1
#else
#define CDNA5_TDM 0
#endif

// 16-bit fragment per-lane layout (ISA 7.12.2): element e -> k:
//   e in [0,8):  k = 8*half + e          (contiguous)
//   e in [8,16): k = 16 + 8*half + (e-8) (contiguous)
// => a fragment is two 16-byte contiguous loads when memory is k-major.
union Frag16 {
  v16bf f;
  struct { v8bf lo; v8bf hi; } h;
};

__device__ __forceinline__ Frag16 load_frag(const __bf16* base_k0, int half) {
  Frag16 r;
  r.h.lo = *(const v8bf*)(base_k0 + 8 * half);
  r.h.hi = *(const v8bf*)(base_k0 + 16 + 8 * half);
  return r;
}

// Stage a 64(row) x 32(k) bf16 tile of Bt[N][K] into LDS buf[64][32].
// TDM path: wave 0 issues one tensor_load_to_lds descriptor (D# per ISA 08 §8):
//   group0: count=1, lds_addr, global_addr, type=2
//   group1: data_size=2B, tile_dim0=32 (k, contiguous), tile_dim1=64 (rows),
//           tensor_dim0_stride=K (data units)
__device__ __forceinline__ void stage_b(__bf16 (*buf)[32], const __bf16* __restrict__ Bt,
                                        int n0, int k0, int K, int wave, int tid) {
#if CDNA5_TDM
  if (wave == 0) {
    uint64_t ga = (uint64_t)(uintptr_t)(Bt + (size_t)n0 * K + k0);
    uint32_t la = (uint32_t)(uintptr_t)(&buf[0][0]);
    unsigned td0 = (unsigned)K;      // tensor_dim0
    unsigned td1 = 65536u;           // tensor_dim1 (large enough)
    u32x4 g0 = { 1u,                                   // count=1, user desc
                 la,                                   // lds_addr
                 (uint32_t)ga,                         // global_addr[31:0]
                 (uint32_t)((ga >> 32) & 0x1FFFFFFu) | 0x80000000u }; // addr[56:32] | type=2
    i32x8 g1 = { (int)0x10000u,                                     // data_size=1 (2 bytes)
                 (int)((td0 & 0xFFFFu) << 16),                      // tensor_dim0 lo16
                 (int)((td0 >> 16) | ((td1 & 0xFFFFu) << 16)),      // td0 hi16 | td1 lo16
                 (int)((td1 >> 16) | (32u << 16)),                  // td1 hi16 | tile_dim0=32
                 (int)64u,                                          // tile_dim1=64, tile_dim2=0
                 (int)(unsigned)K,                                  // dim0_stride lo32
                 0, 0 };
    i32x4 z4 = { 0, 0, 0, 0 };
    i32x8 z8 = { 0, 0, 0, 0, 0, 0, 0, 0 };
    __builtin_amdgcn_tensor_load_to_lds(g0, g1, z4, z4, z8, 0);
  }
#else
  int r = tid >> 2, c = tid & 3;   // 256 threads: 64 rows x 4 chunks of 16B
  *(v8bf*)&buf[r][c * 8] = *(const v8bf*)(Bt + (size_t)(n0 + r) * K + k0 + c * 8);
#endif
}

__device__ __forceinline__ void stage_wait(int wave) {
#if CDNA5_TDM
  if (wave == 0) __builtin_amdgcn_s_wait_tensorcnt(0);
#endif
}

// ---- converts ------------------------------------------------------------
__global__ void f32_to_bf16_kernel(const float* __restrict__ in, __bf16* __restrict__ out, size_t n) {
  size_t i = ((size_t)blockIdx.x * blockDim.x + threadIdx.x) * 8;
  if (i >= n) return;
  float4 a = *(const float4*)(in + i);
  float4 b = *(const float4*)(in + i + 4);
  v8bf o;
  o[0] = (__bf16)a.x; o[1] = (__bf16)a.y; o[2] = (__bf16)a.z; o[3] = (__bf16)a.w;
  o[4] = (__bf16)b.x; o[5] = (__bf16)b.y; o[6] = (__bf16)b.z; o[7] = (__bf16)b.w;
  *(v8bf*)(out + i) = o;
}

// in[R][C] f32 -> out[C][R] bf16  (weights: [K][N] -> [N][K] for k-contiguous B frags)
__global__ void f32_to_bf16_transpose(const float* __restrict__ in, __bf16* __restrict__ out,
                                      int R, int C) {
  int idx = blockIdx.x * blockDim.x + threadIdx.x;
  if (idx >= R * C) return;
  int r = idx / C, c = idx % C;
  out[(size_t)c * R + r] = (__bf16)in[(size_t)r * C + c];
}

// ---- GEMM: C[M,N] = A[M,K] x Bt[N,K]^T + bias, then * scale ---------------
// B tile staged to LDS via TDM, double-buffered (two 32-k steps per loop).
// mode 0: bf16 head-split [b][h][t][dh]
// mode 1: f32 row-major [m][n]
// mode 2: bf16 transposed head-split [b][h][dh][t]   (for V)
__global__ void gemm_bf16_wmma(const __bf16* __restrict__ A, const __bf16* __restrict__ Bt,
                               const float* __restrict__ bias, float scale, int mode,
                               float* __restrict__ outf, __bf16* __restrict__ outh,
                               int M, int N, int K) {
  __shared__ __align__(16) __bf16 bbuf[2][64][32];
  int tid = threadIdx.x;
  int lane = tid & 31, wave = tid >> 5;
  int half = lane >> 4, ln = lane & 15;
  int m0 = blockIdx.x * 128 + wave * 16;
  int n0 = blockIdx.y * 64;
  const __bf16* arow = A + (size_t)(m0 + ln) * K;

  stage_b(bbuf[0], Bt, n0, 0, K, wave, tid);
  stage_wait(wave);
  __syncthreads();

  v8f acc[4] = {};
  Frag16 a_e = load_frag(arow, half);
  Frag16 a_o = load_frag(arow + 32, half);

  for (int k0 = 0; k0 < K; k0 += 64) {
    // ---- even step: compute on bbuf[0], prefetch k0+32 into bbuf[1]
    stage_b(bbuf[1], Bt, n0, k0 + 32, K, wave, tid);
#pragma unroll
    for (int nf = 0; nf < 4; ++nf) {
      Frag16 b = load_frag(&bbuf[0][nf * 16 + ln][0], half);
      acc[nf] = __builtin_amdgcn_wmma_f32_16x16x32_bf16(false, a_e.f, false, b.f,
                                                        (short)0, acc[nf], false, false);
    }
    Frag16 a_e_n = a_e;
    if (k0 + 64 < K) a_e_n = load_frag(arow + k0 + 64, half);
    stage_wait(wave);
    __syncthreads();

    // ---- odd step: compute on bbuf[1], prefetch k0+64 into bbuf[0]
    if (k0 + 64 < K) stage_b(bbuf[0], Bt, n0, k0 + 64, K, wave, tid);
#pragma unroll
    for (int nf = 0; nf < 4; ++nf) {
      Frag16 b = load_frag(&bbuf[1][nf * 16 + ln][0], half);
      acc[nf] = __builtin_amdgcn_wmma_f32_16x16x32_bf16(false, a_o.f, false, b.f,
                                                        (short)0, acc[nf], false, false);
    }
    Frag16 a_o_n = a_o;
    if (k0 + 96 < K) a_o_n = load_frag(arow + k0 + 96, half);
    stage_wait(wave);
    __syncthreads();
    a_e = a_e_n; a_o = a_o_n;
  }

#pragma unroll
  for (int nf = 0; nf < 4; ++nf) {
#pragma unroll
    for (int i = 0; i < 8; ++i) {
      int m = m0 + i + 8 * half;
      int n = n0 + nf * 16 + ln;
      float v = (acc[nf][i] + bias[n]) * scale;
      if (mode == 1) {
        outf[(size_t)m * N + n] = v;
      } else {
        int bidx = m >> 11, t = m & (TSEQ - 1);
        int hh = n >> 6, dh = n & (DHEAD - 1);
        size_t bh = (size_t)bidx * NHEADS + hh;
        if (mode == 0) outh[(bh * TSEQ + t) * DHEAD + dh] = (__bf16)v;
        else           outh[(bh * DHEAD + dh) * TSEQ + t] = (__bf16)v;  // mode 2
      }
    }
  }
}

// qrel[bh][t][r] = dot(q_scaled[bh][t][:], rel_keys[r][:])
__global__ void qrel_kernel(const __bf16* __restrict__ Q, const float* __restrict__ relk,
                            float* __restrict__ qrel, int total) {
  int idx = blockIdx.x * blockDim.x + threadIdx.x;
  if (idx >= total) return;
  int r = idx % NREL;
  int t = (idx / NREL) % TSEQ;
  int bh = idx / (NREL * TSEQ);
  const __bf16* q = Q + ((size_t)bh * TSEQ + t) * DHEAD;
  const float* rk = relk + r * DHEAD;
  float s = 0.f;
#pragma unroll
  for (int d0 = 0; d0 < DHEAD; d0 += 8) {
    v8bf qv = *(const v8bf*)(q + d0);
#pragma unroll
    for (int j = 0; j < 8; ++j) s += (float)qv[j] * rk[d0 + j];
  }
  qrel[idx] = s;
}

// ---- flash attention with relative positions ------------------------------
// grid(T/128, B*H), block 256; each wave owns 16 query rows, key tiles of 32.
// V supplied transposed: Vt[bh][d][T].
__global__ void attn_wmma(const __bf16* __restrict__ Q, const __bf16* __restrict__ Kb,
                          const __bf16* __restrict__ Vt, const float* __restrict__ qrel,
                          const unsigned char* __restrict__ mask,
                          const float* __restrict__ relv, __bf16* __restrict__ headsb) {
  __shared__ __align__(16) float  s_tile[8][16][32];
  __shared__ __align__(16) __bf16 p_tile[8][16][32];
  __shared__ float  asum[8][16][48];   // 41 buckets, padded
  __shared__ float  rowfac[8][16];
  __shared__ float  rowinv[8][16];

  int lane = threadIdx.x & 31, wave = threadIdx.x >> 5;
  int half = lane >> 4, ln = lane & 15;
  int bh = blockIdx.y;
  int b = bh >> 4, h = bh & (NHEADS - 1);
  int qbase = blockIdx.x * 128 + wave * 16;

  { // zero bucket sums
    float* ap = &asum[wave][0][0];
    for (int r = lane; r < 16 * 48; r += 32) ap[r] = 0.f;
  }

  // Q A-fragments (k=0..31, 32..63); 1/sqrt(Dh) already folded into Q
  const __bf16* qrow = Q + (((size_t)bh * TSEQ) + qbase + ln) * DHEAD;
  Frag16 aq0 = load_frag(qrow, half);
  Frag16 aq1 = load_frag(qrow + 32, half);

  v8f o[4] = {};
  float mrow = NEG_BIG;   // lanes 0..15 own row == lane

  for (int s0 = 0; s0 < TSEQ; s0 += 32) {
    if (s0 + 32 < TSEQ)
      __builtin_prefetch(Kb + ((size_t)bh * TSEQ + s0 + 32) * DHEAD, 0, 1);

    // S = Q * K^T (two 16-key column groups, k split 0..31 / 32..63)
    v8f S[2] = {};
#pragma unroll
    for (int c = 0; c < 2; ++c) {
      const __bf16* krow = Kb + ((size_t)bh * TSEQ + s0 + c * 16 + ln) * DHEAD;
      Frag16 b0 = load_frag(krow, half);
      Frag16 b1 = load_frag(krow + 32, half);
      S[c] = __builtin_amdgcn_wmma_f32_16x16x32_bf16(false, aq0.f, false, b0.f, (short)0, S[c], false, false);
      S[c] = __builtin_amdgcn_wmma_f32_16x16x32_bf16(false, aq1.f, false, b1.f, (short)0, S[c], false, false);
    }

    // V B-frags issued early so the loads overlap the softmax bookkeeping
    Frag16 bvv[4];
#pragma unroll
    for (int nf = 0; nf < 4; ++nf)
      bvv[nf] = load_frag(Vt + ((size_t)bh * DHEAD + nf * 16 + ln) * TSEQ + s0, half);

    // rel-key term + mask -> LDS (C layout: elem i -> m=i+8*half, n=ln)
#pragma unroll
    for (int c = 0; c < 2; ++c) {
#pragma unroll
      for (int i = 0; i < 8; ++i) {
        int m = i + 8 * half;
        int t = qbase + m;
        int s = s0 + c * 16 + ln;
        int d = s - t; d = d > 20 ? 20 : (d < -20 ? -20 : d);
        float v = S[c][i] + qrel[((size_t)bh * TSEQ + t) * NREL + (d + 20)];
        if (!mask[b * TSEQ + s]) v = NEG_BIG;
        s_tile[wave][m][c * 16 + ln] = v;
      }
    }
    __syncthreads();

    // per-row online-softmax bookkeeping (lane == row owner)
    if (lane < 16) {
      int row = lane;
      float tmax = NEG_BIG;
#pragma unroll 8
      for (int c = 0; c < 32; ++c) tmax = fmaxf(tmax, s_tile[wave][row][c]);
      float newm = fmaxf(mrow, tmax);
      float fac = __expf(mrow - newm);
      mrow = newm;
      rowfac[wave][row] = fac;
#pragma unroll 8
      for (int r = 0; r < NREL; ++r) asum[wave][row][r] *= fac;
      int t = qbase + row;
#pragma unroll 8
      for (int c = 0; c < 32; ++c) {
        float p = __expf(s_tile[wave][row][c] - newm);
        p_tile[wave][row][c] = (__bf16)p;
        int s = s0 + c;
        int d = s - t; d = d > 20 ? 20 : (d < -20 ? -20 : d);
        asum[wave][row][d + 20] += p;
      }
    }
    __syncthreads();

    // rescale running output
    float fct[8];
#pragma unroll
    for (int i = 0; i < 8; ++i) fct[i] = rowfac[wave][i + 8 * half];
#pragma unroll
    for (int nf = 0; nf < 4; ++nf)
#pragma unroll
      for (int i = 0; i < 8; ++i) o[nf][i] *= fct[i];

    // P A-frag from LDS (row-contiguous -> 2 LDS b128 loads)
    Frag16 pa = load_frag(&p_tile[wave][ln][0], half);
#pragma unroll
    for (int nf = 0; nf < 4; ++nf)
      o[nf] = __builtin_amdgcn_wmma_f32_16x16x32_bf16(false, pa.f, false, bvv[nf].f, (short)0, o[nf], false, false);
    __syncthreads();
  }

  // finalize: l = sum of buckets; heads = (O + asum*rel_values) / l
  if (lane < 16) {
    int row = lane;
    float l = 0.f;
#pragma unroll 8
    for (int r = 0; r < NREL; ++r) l += asum[wave][row][r];
    rowinv[wave][row] = 1.0f / l;
  }
  __syncthreads();

#pragma unroll
  for (int nf = 0; nf < 4; ++nf) {
#pragma unroll
    for (int i = 0; i < 8; ++i) {
      int m = i + 8 * half;
      int t = qbase + m;
      int d = nf * 16 + ln;
      float rv = 0.f;
#pragma unroll 8
      for (int r = 0; r < NREL; ++r) rv += asum[wave][m][r] * relv[r * DHEAD + d];
      float val = (o[nf][i] + rv) * rowinv[wave][m];
      headsb[(((size_t)b * TSEQ) + t) * NUNITS + h * DHEAD + d] = (__bf16)val;
    }
  }
}

extern "C" void kernel_launch(void* const* d_in, const int* in_sizes, int n_in,
                              void* d_out, int out_size, void* d_ws, size_t ws_size,
                              hipStream_t stream) {
  const float* X   = (const float*)d_in[0];
  const unsigned char* mask = (const unsigned char*)d_in[1];
  const float* Wq  = (const float*)d_in[2];
  const float* bq  = (const float*)d_in[3];
  const float* Wk  = (const float*)d_in[4];
  const float* bk  = (const float*)d_in[5];
  const float* Wv  = (const float*)d_in[6];
  const float* bv  = (const float*)d_in[7];
  const float* Wo  = (const float*)d_in[8];
  const float* bo  = (const float*)d_in[9];
  const float* relk = (const float*)d_in[10];
  const float* relv = (const float*)d_in[11];

  char* ws = (char*)d_ws;
  size_t off = 0;
  __bf16* Xb   = (__bf16*)(ws + off); off += (size_t)BROWS * NUNITS * 2;   // 8 MB
  __bf16* Wqt  = (__bf16*)(ws + off); off += (size_t)NUNITS * NUNITS * 2;  // [N][K]
  __bf16* Wkt  = (__bf16*)(ws + off); off += (size_t)NUNITS * NUNITS * 2;
  __bf16* Wvt  = (__bf16*)(ws + off); off += (size_t)NUNITS * NUNITS * 2;
  __bf16* Wot  = (__bf16*)(ws + off); off += (size_t)NUNITS * NUNITS * 2;
  __bf16* Qb   = (__bf16*)(ws + off); off += (size_t)BROWS * NUNITS * 2;   // [b][h][t][dh]
  __bf16* Kbp  = (__bf16*)(ws + off); off += (size_t)BROWS * NUNITS * 2;   // [b][h][t][dh]
  __bf16* Vtp  = (__bf16*)(ws + off); off += (size_t)BROWS * NUNITS * 2;   // [b][h][dh][t]
  float*  qrel = (float*)(ws + off);  off += (size_t)2 * NHEADS * TSEQ * NREL * 4;
  __bf16* headsb = (__bf16*)(ws + off); off += (size_t)BROWS * NUNITS * 2; // [b][t][h*dh]

  const size_t nX = (size_t)BROWS * NUNITS;
  const int    nW = NUNITS * NUNITS;
  f32_to_bf16_kernel<<<(nX / 8 + 255) / 256, 256, 0, stream>>>(X, Xb, nX);
  f32_to_bf16_transpose<<<(nW + 255) / 256, 256, 0, stream>>>(Wq, Wqt, NUNITS, NUNITS);
  f32_to_bf16_transpose<<<(nW + 255) / 256, 256, 0, stream>>>(Wk, Wkt, NUNITS, NUNITS);
  f32_to_bf16_transpose<<<(nW + 255) / 256, 256, 0, stream>>>(Wv, Wvt, NUNITS, NUNITS);
  f32_to_bf16_transpose<<<(nW + 255) / 256, 256, 0, stream>>>(Wo, Wot, NUNITS, NUNITS);

  dim3 gg(BROWS / 128, NUNITS / 64);
  const float qscale = 0.125f; // 1/sqrt(64)
  gemm_bf16_wmma<<<gg, 256, 0, stream>>>(Xb, Wqt, bq, qscale, 0, nullptr, Qb,  BROWS, NUNITS, NUNITS);
  gemm_bf16_wmma<<<gg, 256, 0, stream>>>(Xb, Wkt, bk, 1.0f,   0, nullptr, Kbp, BROWS, NUNITS, NUNITS);
  gemm_bf16_wmma<<<gg, 256, 0, stream>>>(Xb, Wvt, bv, 1.0f,   2, nullptr, Vtp, BROWS, NUNITS, NUNITS);

  int qrel_total = 2 * NHEADS * TSEQ * NREL;
  qrel_kernel<<<(qrel_total + 255) / 256, 256, 0, stream>>>(Qb, relk, qrel, qrel_total);

  attn_wmma<<<dim3(TSEQ / 128, 2 * NHEADS), 256, 0, stream>>>(Qb, Kbp, Vtp, qrel, mask, relv, headsb);

  gemm_bf16_wmma<<<gg, 256, 0, stream>>>(headsb, Wot, bo, 1.0f, 1, (float*)d_out, nullptr, BROWS, NUNITS, NUNITS);
}
